// resnext_block_b_6339371729594
// MI455X (gfx1250) — matrix operand order repaired and checked by reading, not compiled, not run on hardware
//
#include <hip/hip_runtime.h>

typedef __attribute__((ext_vector_type(16))) _Float16 v16h;
typedef __attribute__((ext_vector_type(8)))  float    v8f;

#define EPSBN 1e-3f
#define NPIX  100352   // 32*56*56

// K offset for 16-bit A fragment: VGPR v (0..7), half = lane>>4.
// Per ISA 16x32 f16 A layout: V0:K0,1|K8,9  V1:K2,3|K10,11 ... V4..7 += 16.
__device__ __forceinline__ int ka_map(int v, int half) {
    return ((v & 4) ? 16 : 0) + ((v & 3) << 1) + (half << 3);
}

// ---------------- conv1: 1x1 GEMM 256->128 + bias + BN + ReLU -> f16 ----------------
// grid (784, 2): x = M tiles of 128 pixels (8 waves x 16 rows), y = N half (64 ch)
// Weights staged in LDS pre-swizzled into B-fragment layout:
//   fragment f = ks*4+nt (32 frags), lane l holds 16 contiguous halfs:
//   half q  <->  w[(ks*32 + ((q&8)<<1) + (q&7) + ((l>>4)<<3)) * 128 + nbase + nt*16 + (l&15)]
__global__ __launch_bounds__(256)
void k_conv1(const float* __restrict__ x, const float* __restrict__ w1,
             const float* __restrict__ b1, const float* __restrict__ g1,
             const float* __restrict__ be1, const float* __restrict__ m1,
             const float* __restrict__ v1, _Float16* __restrict__ h1)
{
    __shared__ v16h sWf[32 * 32];              // 32 KB, fragment-major, 32B/lane
    _Float16* sW = (_Float16*)sWf;
    const int tid   = threadIdx.x;
    const int nbase = blockIdx.y * 64;
    for (int i = tid; i < 32 * 512; i += 256) {
        int f = i >> 9;                        // ks*4 + nt
        int l = (i >> 4) & 31;                 // lane
        int q = i & 15;                        // half within lane
        int ks = f >> 2, nt = f & 3;
        int k = ks * 32 + ((q & 8) << 1) + (q & 7) + ((l >> 4) << 3);
        int n = nt * 16 + (l & 15);
        sW[i] = (_Float16)w1[k * 128 + nbase + n];
    }
    __syncthreads();

    const int lane = tid & 31;
    const int wave = tid >> 5;
    const int half = lane >> 4;
    const int l16  = lane & 15;
    const int m0   = (blockIdx.x * 8 + wave) * 16;

    v8f zero = {};
    v8f acc[4];
#pragma unroll
    for (int t = 0; t < 4; ++t) acc[t] = zero;

    const float* xrow = x + (long)(m0 + l16) * 256;
#pragma unroll
    for (int ks = 0; ks < 8; ++ks) {
        const int kb = ks * 32;
        v16h a;
#pragma unroll
        for (int v = 0; v < 8; ++v) {
            int k = kb + ka_map(v, half);
            a[2 * v]     = (_Float16)xrow[k];
            a[2 * v + 1] = (_Float16)xrow[k + 1];
        }
#pragma unroll
        for (int nt = 0; nt < 4; ++nt) {
            v16h b = sWf[(ks * 4 + nt) * 32 + lane];   // 2x ds_load_b128
            acc[nt] = __builtin_amdgcn_wmma_f32_16x16x32_f16(
                false, a, false, b, (short)0, acc[nt], false, false);
        }
    }
    // fused bias + BN + ReLU, store f16
#pragma unroll
    for (int nt = 0; nt < 4; ++nt) {
        int n = nbase + nt * 16 + l16;
        float s = g1[n] * rsqrtf(v1[n] + EPSBN);
        float t = (b1[n] - m1[n]) * s + be1[n];
#pragma unroll
        for (int r = 0; r < 8; ++r) {
            int m = m0 + r + half * 8;
            float val = acc[nt][r] * s + t;
            val = val > 0.f ? val : 0.f;
            h1[(long)m * 128 + n] = (_Float16)val;
        }
    }
}

// ------------- conv2: grouped 3x3 (32 groups of 4->4) + bias + BN + ReLU -------------
// one thread per (pixel, group); wave = 1 pixel x 32 groups (coalesced 8B/lane loads)
__global__ __launch_bounds__(256)
void k_conv2(const _Float16* __restrict__ h1, const float* __restrict__ w2,
             const float* __restrict__ b2, const float* __restrict__ g2,
             const float* __restrict__ be2, const float* __restrict__ m2,
             const float* __restrict__ v2, _Float16* __restrict__ h2)
{
    int idx = blockIdx.x * 256 + threadIdx.x;  // pixel*32 + group
    int g = idx & 31;
    int p = idx >> 5;
    int ww = p % 56;
    int t0 = p / 56;
    int hh = t0 % 56;
    int bb = t0 / 56;

    float acc[4] = {0.f, 0.f, 0.f, 0.f};
#pragma unroll
    for (int ky = 0; ky < 3; ++ky) {
        int y = hh + ky - 1;
        if (y < 0 || y >= 56) continue;
#pragma unroll
        for (int kx = 0; kx < 3; ++kx) {
            int xx = ww + kx - 1;
            if (xx < 0 || xx >= 56) continue;
            const _Float16* in = h1 + ((long)((bb * 56 + y) * 56 + xx)) * 128 + g * 4;
            float i0 = (float)in[0], i1 = (float)in[1];
            float i2 = (float)in[2], i3 = (float)in[3];
            const float* wp = w2 + (ky * 3 + kx) * 4 * 128 + g * 4;  // [tap][ic][128]
#pragma unroll
            for (int oc = 0; oc < 4; ++oc)
                acc[oc] += i0 * wp[0 * 128 + oc] + i1 * wp[1 * 128 + oc]
                         + i2 * wp[2 * 128 + oc] + i3 * wp[3 * 128 + oc];
        }
    }
#pragma unroll
    for (int oc = 0; oc < 4; ++oc) {
        int n = g * 4 + oc;
        float s = g2[n] * rsqrtf(v2[n] + EPSBN);
        float t = (b2[n] - m2[n]) * s + be2[n];
        float val = acc[oc] * s + t;
        val = val > 0.f ? val : 0.f;
        h2[(long)p * 128 + n] = (_Float16)val;
    }
}

// -------- conv3: 1x1 GEMM 128->256 + bias + residual(x) + ReLU -> f32 out --------
// grid (784, 2): y = N half (128 ch); 8 N-tiles per wave, K=128 (4 k-steps)
// Same pre-swizzled B-fragment LDS layout: fragment f = ks*8+nt (32 frags).
__global__ __launch_bounds__(256)
void k_conv3(const _Float16* __restrict__ h2, const float* __restrict__ w3,
             const float* __restrict__ b3, const float* __restrict__ x,
             float* __restrict__ out)
{
    __shared__ v16h sWf[32 * 32];              // 32 KB
    _Float16* sW = (_Float16*)sWf;
    const int tid   = threadIdx.x;
    const int nbase = blockIdx.y * 128;
    for (int i = tid; i < 32 * 512; i += 256) {
        int f = i >> 9;                        // ks*8 + nt
        int l = (i >> 4) & 31;
        int q = i & 15;
        int ks = f >> 3, nt = f & 7;
        int k = ks * 32 + ((q & 8) << 1) + (q & 7) + ((l >> 4) << 3);
        int n = nt * 16 + (l & 15);
        sW[i] = (_Float16)w3[k * 256 + nbase + n];
    }
    __syncthreads();

    const int lane = tid & 31;
    const int wave = tid >> 5;
    const int half = lane >> 4;
    const int l16  = lane & 15;
    const int m0   = (blockIdx.x * 8 + wave) * 16;

    // warm the residual row into cache for the epilogue
    __builtin_prefetch(x + (long)(m0 + l16) * 256 + nbase, 0, 1);

    v8f zero = {};
    v8f acc[8];
#pragma unroll
    for (int t = 0; t < 8; ++t) acc[t] = zero;

    const _Float16* arow = h2 + (long)(m0 + l16) * 128;
#pragma unroll
    for (int ks = 0; ks < 4; ++ks) {
        const int kb = ks * 32;
        v16h a;
#pragma unroll
        for (int v = 0; v < 8; ++v) {
            int k = kb + ka_map(v, half);
            a[2 * v]     = arow[k];
            a[2 * v + 1] = arow[k + 1];
        }
#pragma unroll
        for (int nt = 0; nt < 8; ++nt) {
            v16h b = sWf[(ks * 8 + nt) * 32 + lane];   // 2x ds_load_b128
            acc[nt] = __builtin_amdgcn_wmma_f32_16x16x32_f16(
                false, a, false, b, (short)0, acc[nt], false, false);
        }
    }
    // bias + residual + ReLU, fp32 out
#pragma unroll
    for (int nt = 0; nt < 8; ++nt) {
        int n = nbase + nt * 16 + l16;
        float bias = b3[n];
#pragma unroll
        for (int r = 0; r < 8; ++r) {
            int m = m0 + r + half * 8;
            float val = acc[nt][r] + bias + x[(long)m * 256 + n];
            val = val > 0.f ? val : 0.f;
            out[(long)m * 256 + n] = val;
        }
    }
}

extern "C" void kernel_launch(void* const* d_in, const int* in_sizes, int n_in,
                              void* d_out, int out_size, void* d_ws, size_t ws_size,
                              hipStream_t stream) {
    const float* x   = (const float*)d_in[0];
    const float* w1  = (const float*)d_in[1];
    const float* b1  = (const float*)d_in[2];
    const float* g1  = (const float*)d_in[3];
    const float* be1 = (const float*)d_in[4];
    const float* m1  = (const float*)d_in[5];
    const float* v1  = (const float*)d_in[6];
    const float* w2  = (const float*)d_in[7];
    const float* b2  = (const float*)d_in[8];
    const float* g2  = (const float*)d_in[9];
    const float* be2 = (const float*)d_in[10];
    const float* m2  = (const float*)d_in[11];
    const float* v2  = (const float*)d_in[12];
    const float* w3  = (const float*)d_in[13];
    const float* b3  = (const float*)d_in[14];
    float* out = (float*)d_out;

    _Float16* h1 = (_Float16*)d_ws;                  // 100352*128 f16 = 25.7 MB
    _Float16* h2 = h1 + (size_t)NPIX * 128;          // another 25.7 MB

    dim3 blk(256);
    k_conv1<<<dim3(784, 2),  blk, 0, stream>>>(x, w1, b1, g1, be1, m1, v1, h1);
    k_conv2<<<dim3(12544),   blk, 0, stream>>>(h1, w2, b2, g2, be2, m2, v2, h2);
    k_conv3<<<dim3(784, 2),  blk, 0, stream>>>(h2, w3, b3, x, out);
}
